// INP_MedClassification_52750788329760
// MI455X (gfx1250) — matrix operand order repaired and checked by reading, not compiled, not run on hardware
//
#include <hip/hip_runtime.h>
#include <hip/hip_bf16.h>
#include <math.h>

typedef __bf16 bf16_t;
typedef bf16_t v16bf __attribute__((ext_vector_type(16)));
typedef float  v8f   __attribute__((ext_vector_type(8)));
typedef unsigned int uint4v __attribute__((ext_vector_type(4)));

#define BS_ 64
#define M_ 1000
#define D_ 512
#define NWAYS_ 20
#define NZ_ 8

// ---------------- fragment loaders ----------------
// A fragment (16x32 bf16): lane m holds K {0..7,16..23}; lane m+16 holds K {8..15,24..31}
__device__ __forceinline__ v16bf load_frag_a_bf(const bf16_t* p, int k0) {
  union { uint4v u[2]; v16bf v; } r;
  r.u[0] = *(const uint4v*)(p + k0);
  r.u[1] = *(const uint4v*)(p + k0 + 16);
  return r.v;
}
__device__ __forceinline__ v16bf load_frag_a_f32(const float* p, int k0) {
  v16bf r;
#pragma unroll
  for (int i = 0; i < 8; ++i) r[i] = (bf16_t)p[k0 + i];
#pragma unroll
  for (int i = 0; i < 8; ++i) r[8 + i] = (bf16_t)p[k0 + 16 + i];
  return r;
}
// B fragment (32x16 bf16) from row n of B^T: lane n holds K {0..7,16..23}, lane n+16 holds K {8..15,24..31}
__device__ __forceinline__ v16bf load_frag_b(const bf16_t* p, int kb) {
  union { uint4v u[2]; v16bf v; } r;
  r.u[0] = *(const uint4v*)(p + kb);
  r.u[1] = *(const uint4v*)(p + kb + 16);
  return r.v;
}

__device__ __forceinline__ float gelu_exact(float x) {
  return 0.5f * x * (1.0f + erff(x * 0.70710678118654752f));
}
__device__ __forceinline__ float softplus_f(float x) {
  return (x > 20.0f) ? x : log1pf(expf(x));
}

__device__ __forceinline__ v8f wmma_bf16(v16bf a, v16bf b, v8f c) {
  return __builtin_amdgcn_wmma_f32_16x16x32_bf16(false, a, false, b, (short)0, c, false, false);
}

// ---------------- pipelined bf16 WMMA GEMM ----------------
// C[M,N] = epi(A[M,K] @ B[K,N] + bias), B given as BT[N,K] bf16 row-major.
// grid = (M/128, N/64), block = 128 (4 waves); wave w -> 32 rows x 64 cols.
// Double-buffered: next A frags + next B frag loaded before current WMMA pair.
// EPI: 0 = bias -> bf16 out; 1 = gelu(bias) -> bf16 out; 2 = stats split (mu / scale, f32).
template <int AF32, int EPI>
__global__ void gemm_bf16_kernel(const void* __restrict__ Av,
                                 const bf16_t* __restrict__ BT,
                                 const float* __restrict__ bias,
                                 bf16_t* __restrict__ out_bf,
                                 float* __restrict__ out_mu,
                                 float* __restrict__ out_scale,
                                 int K, int N) {
  const int wave = threadIdx.x >> 5;
  const int lane = threadIdx.x & 31;
  const int ncol = lane & 15;
  const int k0a  = (lane < 16) ? 0 : 8;
  const int kb   = (lane < 16) ? 0 : 8;
  const size_t row0 = (size_t)blockIdx.x * 128 + wave * 32;   // 32 rows per wave
  const int colBase = blockIdx.y * 64;

  const v8f zero = {0.f, 0.f, 0.f, 0.f, 0.f, 0.f, 0.f, 0.f};
  v8f acc0[4] = {zero, zero, zero, zero};   // rows row0..row0+15
  v8f acc1[4] = {zero, zero, zero, zero};   // rows row0+16..row0+31

  const float*  Af = (const float*)Av;
  const bf16_t* Ab = (const bf16_t*)Av;
  const size_t arow0 = (row0 + (size_t)ncol) * (size_t)K;
  const size_t arow1 = (row0 + 16 + (size_t)ncol) * (size_t)K;
  const bf16_t* brow = BT + (size_t)(colBase + ncol) * K;   // col tile j at + j*16*K

  // prologue: k = 0 fragments
  v16bf a0, a1, b;
  if (AF32) { a0 = load_frag_a_f32(Af + arow0, k0a); a1 = load_frag_a_f32(Af + arow1, k0a); }
  else      { a0 = load_frag_a_bf(Ab + arow0, k0a);  a1 = load_frag_a_bf(Ab + arow1, k0a); }
  b = load_frag_b(brow, kb);

  for (int k = 0; k < K; k += 32) {
    const int kn = (k + 32 < K) ? (k + 32) : 0;   // wrapped (dummy) prefetch on last iter
    // prefetch next-k A fragments early
    v16bf a0n, a1n;
    if (AF32) { a0n = load_frag_a_f32(Af + arow0 + kn, k0a); a1n = load_frag_a_f32(Af + arow1 + kn, k0a); }
    else      { a0n = load_frag_a_bf(Ab + arow0 + kn, k0a);  a1n = load_frag_a_bf(Ab + arow1 + kn, k0a); }
#pragma unroll
    for (int j = 0; j < 4; ++j) {
      // prefetch next B fragment (next col tile, or col tile 0 of next k)
      const bf16_t* bnp = (j < 3) ? (brow + (size_t)(j + 1) * 16 * K + k)
                                  : (brow + kn);
      v16bf bn = load_frag_b(bnp, kb);
      acc0[j] = wmma_bf16(a0, b, acc0[j]);
      acc1[j] = wmma_bf16(a1, b, acc1[j]);
      b = bn;
    }
    a0 = a0n; a1 = a1n;
  }

  const int lh = lane >> 4;
#pragma unroll
  for (int j = 0; j < 4; ++j) {
    const int col = colBase + j * 16 + ncol;
    const float bv = bias[col];
#pragma unroll
    for (int t = 0; t < 2; ++t) {
      const v8f* acc = t ? acc1 : acc0;
#pragma unroll
      for (int v = 0; v < 8; ++v) {
        const size_t m = row0 + t * 16 + v + lh * 8;
        float val = acc[j][v] + bv;
        if (EPI == 2) {
          if (col < 512) out_mu[m * 512 + col] = val;
          else           out_scale[m * 512 + (col - 512)] = 0.01f + 0.99f * softplus_f(val);
        } else {
          if (EPI == 1) val = gelu_exact(val);
          out_bf[m * (size_t)N + col] = (bf16_t)val;
        }
      }
    }
  }
}

// ---------------- logits GEMM: per (s,b), 1000x512 @ (512x20) ----------------
// grid (16, 512): blockIdx.y = s*64+b; block 128; wave handles 16 rows x 32 cols (2 tiles).
__global__ void logits_kernel(const bf16_t* __restrict__ xq,
                              const bf16_t* __restrict__ Wmat,
                              float* __restrict__ logits) {
  const int sb   = blockIdx.y;
  const int wave = threadIdx.x >> 5;
  const int lane = threadIdx.x & 31;
  const int mtile = blockIdx.x * 4 + wave;     // 0..63 (tile 63 is all-clamped)
  const int bidx = sb & 63;
  const bf16_t* A  = xq + (size_t)bidx * M_ * D_;
  const bf16_t* BT = Wmat + (size_t)sb * NWAYS_ * D_;

  int row = mtile * 16 + (lane & 15);
  if (row > M_ - 1) row = M_ - 1;             // clamp loads; stores masked below
  const int k0a = (lane < 16) ? 0 : 8;
  const int kb  = (lane < 16) ? 0 : 8;
  const int ncol = lane & 15;
  const int n0 = ncol;                         // always < 20
  const int n1 = 16 + ncol;
  const int n1c = (n1 < NWAYS_) ? n1 : 0;      // clamp B row for n>=20

  const v8f zero = {0.f, 0.f, 0.f, 0.f, 0.f, 0.f, 0.f, 0.f};
  v8f c0 = zero, c1 = zero;
  const bf16_t* arow = A + (size_t)row * D_;
  const bf16_t* b0p  = BT + (size_t)n0 * D_;
  const bf16_t* b1p  = BT + (size_t)n1c * D_;

  v16bf a  = load_frag_a_bf(arow, k0a);
  v16bf b0 = load_frag_b(b0p, kb);
  v16bf b1 = load_frag_b(b1p, kb);
  for (int k = 0; k < D_; k += 32) {
    const int kn = (k + 32 < D_) ? (k + 32) : 0;
    v16bf an  = load_frag_a_bf(arow + kn, k0a);
    v16bf b0n = load_frag_b(b0p + kn, kb);
    v16bf b1n = load_frag_b(b1p + kn, kb);
    c0 = wmma_bf16(a, b0, c0);
    c1 = wmma_bf16(a, b1, c1);
    a = an; b0 = b0n; b1 = b1n;
  }

  const int lh = lane >> 4;
#pragma unroll
  for (int v = 0; v < 8; ++v) {
    const int m = mtile * 16 + v + lh * 8;
    if (m < M_) {
      float* orow = logits + ((size_t)sb * M_ + m) * NWAYS_;
      orow[n0] = c0[v];
      if (n1 < NWAYS_) orow[n1] = c1[v];
    }
  }
}

// ---------------- weight transpose + bf16 convert ----------------
// BT[n*K + k] = (bf16) W[k*N + n]
__global__ void transpose_w_kernel(const float* __restrict__ W, bf16_t* __restrict__ BT,
                                   int K, int N) {
  const int idx = blockIdx.x * 256 + threadIdx.x;
  if (idx >= K * N) return;
  const int n = idx / K, k = idx - n * K;
  BT[idx] = (bf16_t)W[(size_t)k * N + n];
}

// ---------------- class scatter-reduce + relu: comb[b,c,d] ----------------
// grid (64, 2), block 256. Deterministic (no atomics).
__global__ void reduce_classes_kernel(const bf16_t* __restrict__ xc,
                                      const int* __restrict__ y,
                                      bf16_t* __restrict__ comb) {
  __shared__ int   ycls[M_];
  __shared__ float accs[NWAYS_ * 256];  // [c*256 + tid]
  const int b = blockIdx.x;
  const int tid = threadIdx.x;
  const int d = blockIdx.y * 256 + tid;

  for (int i = tid; i < M_; i += 256) ycls[i] = y[b * M_ + i];
#pragma unroll
  for (int c = 0; c < NWAYS_; ++c) accs[c * 256 + tid] = 0.f;
  __syncthreads();

  const bf16_t* xb = xc + (size_t)b * M_ * D_ + d;
  for (int m = 0; m < M_; ++m) {
    const float v = (float)xb[(size_t)m * D_];
    const int c = ycls[m];
    accs[c * 256 + tid] += v;
  }
#pragma unroll
  for (int c = 0; c < NWAYS_; ++c) {
    const float v = accs[c * 256 + tid];
    comb[((size_t)b * NWAYS_ + c) * D_ + d] = (bf16_t)fmaxf(v, 0.f);
  }
}

// ---------------- z sampling ----------------
__device__ __forceinline__ unsigned pcg_hash(unsigned x) {
  x = x * 747796405u + 2891336453u;
  unsigned w = ((x >> ((x >> 28u) + 4u)) ^ x) * 277803737u;
  return (w >> 22u) ^ w;
}
__device__ __forceinline__ float eps_normal(unsigned i) {
  const unsigned a = pcg_hash(2u * i + 1u);
  const unsigned b = pcg_hash(2u * i + 2u);
  const float u1 = ((float)a + 1.0f) * (1.0f / 4294967296.0f);
  const float u2 = (float)b * (1.0f / 4294967296.0f);
  return sqrtf(-2.0f * logf(u1)) * cosf(6.28318530717958648f * u2);
}
__global__ void gen_z_kernel(const float* __restrict__ mu, const float* __restrict__ scale,
                             float* __restrict__ z_out, bf16_t* __restrict__ z_bf) {
  const int idx = blockIdx.x * 256 + threadIdx.x;  // < 8*1280*512
  const int rem = idx % (BS_ * NWAYS_ * D_);
  const float e = eps_normal((unsigned)idx);
  const float v = mu[rem] + scale[rem] * e;
  z_out[idx] = v;
  z_bf[idx] = (bf16_t)v;
}

// =====================================================================
extern "C" void kernel_launch(void* const* d_in, const int* in_sizes, int n_in,
                              void* d_out, int out_size, void* d_ws, size_t ws_size,
                              hipStream_t stream) {
  const float* x_context = (const float*)d_in[0];
  const int*   y_context = (const int*)d_in[1];
  const float* x_query   = (const float*)d_in[2];
  const float* Wp  = (const float*)d_in[3];
  const float* bp  = (const float*)d_in[4];
  const float* Wa1 = (const float*)d_in[5];
  const float* ba1 = (const float*)d_in[6];
  const float* Wa2 = (const float*)d_in[7];
  const float* ba2 = (const float*)d_in[8];
  const float* Wd1 = (const float*)d_in[9];
  const float* bd1 = (const float*)d_in[10];
  const float* Wd2 = (const float*)d_in[11];
  const float* bd2 = (const float*)d_in[12];

  // output layout: logits | z_samples | mu | scale
  float* logits = (float*)d_out;                                   // 8*64*1000*20
  float* z_out  = logits + (size_t)NZ_ * BS_ * M_ * NWAYS_;        // 8*64*20*512
  float* mu     = z_out + (size_t)NZ_ * BS_ * NWAYS_ * D_;         // 64*20*512
  float* scale  = mu + (size_t)BS_ * NWAYS_ * D_;

  // workspace carve (bf16)
  bf16_t* p = (bf16_t*)d_ws;
  bf16_t* WpT  = p; p += 512 * 512;
  bf16_t* Wa1T = p; p += 512 * 512;
  bf16_t* Wa2T = p; p += 1024 * 512;
  bf16_t* Wd1T = p; p += 512 * 512;
  bf16_t* Wd2T = p; p += 512 * 512;
  bf16_t* xc_bf = p; p += (size_t)BS_ * M_ * D_;      // 64000 x 512
  bf16_t* xq_bf = p; p += (size_t)BS_ * M_ * D_;
  bf16_t* comb  = p; p += (size_t)BS_ * NWAYS_ * D_;  // 1280 x 512
  bf16_t* h_bf  = p; p += (size_t)BS_ * NWAYS_ * D_;
  bf16_t* z_bf  = p; p += (size_t)NZ_ * BS_ * NWAYS_ * D_;  // 10240 x 512
  bf16_t* t_bf  = p; p += (size_t)NZ_ * BS_ * NWAYS_ * D_;
  bf16_t* Wm_bf = p; p += (size_t)NZ_ * BS_ * NWAYS_ * D_;
  (void)ws_size; (void)in_sizes; (void)n_in; (void)out_size;

  // 1) weight transposes -> bf16 B^T
  transpose_w_kernel<<<(512 * 512 + 255) / 256, 256, 0, stream>>>(Wp, WpT, 512, 512);
  transpose_w_kernel<<<(512 * 512 + 255) / 256, 256, 0, stream>>>(Wa1, Wa1T, 512, 512);
  transpose_w_kernel<<<(512 * 1024 + 255) / 256, 256, 0, stream>>>(Wa2, Wa2T, 512, 1024);
  transpose_w_kernel<<<(512 * 512 + 255) / 256, 256, 0, stream>>>(Wd1, Wd1T, 512, 512);
  transpose_w_kernel<<<(512 * 512 + 255) / 256, 256, 0, stream>>>(Wd2, Wd2T, 512, 512);

  // 2) projections: xc / xq (fp32 A -> bf16 out), M = 64000 (500 blocks of 128 rows)
  gemm_bf16_kernel<1, 0><<<dim3(500, 8), 128, 0, stream>>>(x_context, WpT, bp, xc_bf, nullptr, nullptr, 512, 512);
  gemm_bf16_kernel<1, 0><<<dim3(500, 8), 128, 0, stream>>>(x_query,   WpT, bp, xq_bf, nullptr, nullptr, 512, 512);

  // 3) class scatter-reduce + relu -> comb (1280 x 512)
  reduce_classes_kernel<<<dim3(BS_, 2), 256, 0, stream>>>(xc_bf, y_context, comb);

  // 4) h = gelu(comb @ Wa1 + ba1), M = 1280
  gemm_bf16_kernel<0, 1><<<dim3(10, 8), 128, 0, stream>>>(comb, Wa1T, ba1, h_bf, nullptr, nullptr, 512, 512);

  // 5) stats = h @ Wa2 + ba2 -> mu, scale (N = 1024)
  gemm_bf16_kernel<0, 2><<<dim3(10, 16), 128, 0, stream>>>(h_bf, Wa2T, ba2, nullptr, mu, scale, 512, 1024);

  // 6) z = mu + scale * eps  (8*1280*512 elements)
  gen_z_kernel<<<(NZ_ * BS_ * NWAYS_ * D_) / 256, 256, 0, stream>>>(mu, scale, z_out, z_bf);

  // 7) decoder: t = gelu(z @ Wd1 + bd1); Wmat = t @ Wd2 + bd2, M = 10240
  gemm_bf16_kernel<0, 1><<<dim3(80, 8), 128, 0, stream>>>(z_bf, Wd1T, bd1, t_bf, nullptr, nullptr, 512, 512);
  gemm_bf16_kernel<0, 0><<<dim3(80, 8), 128, 0, stream>>>(t_bf, Wd2T, bd2, Wm_bf, nullptr, nullptr, 512, 512);

  // 8) logits[s,b,q,n] = xq[b,q,:] . Wmat[s,b,n,:]
  logits_kernel<<<dim3(16, NZ_ * BS_), 128, 0, stream>>>(xq_bf, Wm_bf, logits);
}
